// PerformerAttention_54503134986263
// MI455X (gfx1250) — matrix operand order repaired and checked by reading, not compiled, not run on hardware
//
#include <hip/hip_runtime.h>
#include <hip/hip_bf16.h>

// ---------------------------------------------------------------------------
// Performer attention for MI455X (gfx1250), bf16 WMMA everywhere.
// B=4, S=4096, D=1024, H=16, R=256, HD=64
// ---------------------------------------------------------------------------

#define Bsz 4
#define Sq  4096
#define Dm  1024
#define Hn  16
#define Rf  256
#define HDm 64
#define BH  (Bsz * Hn)

typedef __attribute__((ext_vector_type(16))) __bf16 v16bf;
typedef __attribute__((ext_vector_type(8)))  float  v8f;

#define WMMA_BF16(a, b, c) \
  __builtin_amdgcn_wmma_f32_16x16x32_bf16(false, (a), false, (b), (short)0, (c), false, false)

// Async global->LDS copy path (gfx1250): gated so compile never regresses.
// Prototype (from hipcc diagnostic): param1 = AS1 pointer to int4 (global src),
// param2 = AS3 pointer to int4 (LDS dst), then imm offset + cpol.
#if defined(__gfx1250__) && \
    __has_builtin(__builtin_amdgcn_global_load_async_to_lds_b128) && \
    __has_builtin(__builtin_amdgcn_s_wait_asynccnt)
#define HAVE_ASYNC_LDS 1
typedef int v4i __attribute__((vector_size(4 * sizeof(int))));
typedef __attribute__((address_space(1))) v4i* g_v4i_p;
typedef __attribute__((address_space(3))) v4i* l_v4i_p;
#else
#define HAVE_ASYNC_LDS 0
#endif

// ----- WMMA fragment loaders (layouts per CDNA5 ISA 7.12.2, wave32) --------
// A (16x32 bf16, MxK): lane m = l&15; lanes 0-15 -> K 0..7 & 16..23,
//                      lanes 16-31 -> K 8..15 & 24..31.
__device__ __forceinline__ v16bf frag_a(const __bf16* base, int ld) {
  const int l  = threadIdx.x & 31;
  const int m  = l & 15;
  const int kb = (l < 16) ? 0 : 8;
  const __bf16* p = base + m * ld + kb;
  v16bf f;
#pragma unroll
  for (int j = 0; j < 8; ++j) { f[j] = p[j]; f[j + 8] = p[j + 16]; }
  return f;
}

// B (32x16 bf16, KxN): lane n = l&15; lanes 0-15 -> K 0..7 & 16..23,
//                      lanes 16-31 -> K 8..15 & 24..31.
__device__ __forceinline__ v16bf frag_b(const __bf16* base, int ld) {
  const int l  = threadIdx.x & 31;
  const int n  = l & 15;
  const int kb = (l < 16) ? 0 : 8;
  const __bf16* p = base + n * ld + kb;
  v16bf f;
#pragma unroll
  for (int j = 0; j < 8; ++j) { f[j] = p[j]; f[j + 8] = p[j + 16]; }
  return f;
}

// ---------------------------------------------------------------------------
// Kernel 0: zero-fill scratch (graph-capture-safe init of atomics targets)
// ---------------------------------------------------------------------------
__global__ void zero_kernel(float* __restrict__ p, int n) {
  int i = blockIdx.x * 256 + threadIdx.x;
  if (i < n) p[i] = 0.0f;
}

// ---------------------------------------------------------------------------
// Kernel 1: C = A[16384,1024] * W[1024,1024]^T + bias   (bf16 WMMA, f32 acc)
// mode 0: C is [M, 1024] f32 (final output)
// mode 1: C is [B,H,S,HD] f32 (head-split projection output)
// Block: 256 thr (8 waves), tile 128x128, wave tile 32x64.
// ---------------------------------------------------------------------------
#define GKP 40  // 32 + 8 pad (bf16 elems)
__global__ __launch_bounds__(256) void gemm_bt_kernel(
    const float* __restrict__ A, const float* __restrict__ W,
    const float* __restrict__ bias, float* __restrict__ C, int mode) {
  __shared__ __bf16 a_sh[128 * GKP];
  __shared__ __bf16 b_sh[128 * GKP];
  const int t   = threadIdx.x;
  const int wid = t >> 5;
  const int l   = t & 31;
  const int m0  = blockIdx.x * 128;
  const int n0  = blockIdx.y * 128;
  const int wm  = (wid >> 1) * 32;
  const int wn  = (wid & 1) * 64;

  v8f acc[2][4] = {};

  const int srow = t >> 1;
  const int sk   = (t & 1) * 16;

  for (int k0 = 0; k0 < Dm; k0 += 32) {
    const float* ap = A + (size_t)(m0 + srow) * Dm + k0 + sk;
    const float* wp = W + (size_t)(n0 + srow) * Dm + k0 + sk;
    // Speculative prefetch of next K tile (safe past-end: silently dropped).
    __builtin_prefetch(ap + 32, 0, 1);   // global_prefetch_b8
    __builtin_prefetch(wp + 32, 0, 1);
#pragma unroll
    for (int i = 0; i < 16; ++i) {
      a_sh[srow * GKP + sk + i] = (__bf16)ap[i];
      b_sh[srow * GKP + sk + i] = (__bf16)wp[i];
    }
    __syncthreads();
    v16bf af[2];
#pragma unroll
    for (int mi = 0; mi < 2; ++mi) af[mi] = frag_a(a_sh + (wm + mi * 16) * GKP, GKP);
#pragma unroll
    for (int nt = 0; nt < 4; ++nt) {
      v16bf bf = frag_b(b_sh + (wn + nt * 16) * GKP, GKP);
#pragma unroll
      for (int mi = 0; mi < 2; ++mi) acc[mi][nt] = WMMA_BF16(af[mi], bf, acc[mi][nt]);
    }
    __syncthreads();
  }

  const int cn = l & 15;
  const int rb = (l < 16) ? 0 : 8;
#pragma unroll
  for (int mi = 0; mi < 2; ++mi) {
#pragma unroll
    for (int nt = 0; nt < 4; ++nt) {
#pragma unroll
      for (int r = 0; r < 8; ++r) {
        const int   mg = m0 + wm + mi * 16 + rb + r;
        const int   ng = n0 + wn + nt * 16 + cn;
        const float v  = acc[mi][nt][r] + bias[ng];
        if (mode == 0) {
          C[(size_t)mg * Dm + ng] = v;
        } else {
          const int b = mg >> 12, s = mg & (Sq - 1);
          const int h = ng >> 6, hd = ng & 63;
          C[(((size_t)(b * Hn + h) * Sq + s) << 6) + hd] = v;
        }
      }
    }
  }
}

// ---------------------------------------------------------------------------
// Kernel 2: feature map  zp = exp(z @ rf[h] - 0.5*||z_row||^2), bf16 out.
// Per (b,h): [4096 x 64] x [64 x 256].  Block tile 128x128, K=64 staged once.
// sumsq computed from f32 staging data (exact), exp in f32 VALU.
// ---------------------------------------------------------------------------
#define FKP 72  // 64 + 8 pad
__global__ __launch_bounds__(256) void feature_kernel(
    const float* __restrict__ Z, const float* __restrict__ RF,
    __bf16* __restrict__ ZP) {
  __shared__ __bf16 a_sh[128 * FKP];
  __shared__ __bf16 b_sh[128 * FKP];
  __shared__ float  part[256];
  __shared__ float  ssq[128];
  const int t   = threadIdx.x;
  const int wid = t >> 5;
  const int l   = t & 31;
  const int s0  = blockIdx.x * 128;
  const int r0  = blockIdx.y * 128;
  const int bh  = blockIdx.z;
  const int h   = bh & (Hn - 1);
  const int wm  = (wid >> 1) * 32;
  const int wn  = (wid & 1) * 64;

  {  // stage A [128 x 64] + partial sumsq (f32)
    const int row = t >> 1, ks = (t & 1) * 32;
    const float* zp = Z + ((size_t)bh * Sq + s0 + row) * HDm + ks;
    float ss = 0.0f;
#pragma unroll
    for (int i = 0; i < 32; ++i) {
      const float v = zp[i];
      ss += v * v;
      a_sh[row * FKP + ks + i] = (__bf16)v;
    }
    part[t] = ss;
  }
  {  // stage B [128 n x 64 k] from RF[h][k][r] (transpose while converting)
    const int k = t >> 2, ns = (t & 3) * 32;
    const float* rp = RF + ((size_t)h * HDm + k) * Rf + r0 + ns;
#pragma unroll
    for (int j = 0; j < 32; ++j) b_sh[(ns + j) * FKP + k] = (__bf16)rp[j];
  }
  __syncthreads();
  if (t < 128) ssq[t] = part[2 * t] + part[2 * t + 1];
  __syncthreads();

  v8f acc[2][4] = {};
#pragma unroll
  for (int kt = 0; kt < 64; kt += 32) {
    v16bf af[2];
#pragma unroll
    for (int mi = 0; mi < 2; ++mi) af[mi] = frag_a(a_sh + (wm + mi * 16) * FKP + kt, FKP);
#pragma unroll
    for (int nt = 0; nt < 4; ++nt) {
      v16bf bf = frag_b(b_sh + (wn + nt * 16) * FKP + kt, FKP);
#pragma unroll
      for (int mi = 0; mi < 2; ++mi) acc[mi][nt] = WMMA_BF16(af[mi], bf, acc[mi][nt]);
    }
  }

  const int cn = l & 15;
  const int rb = (l < 16) ? 0 : 8;
#pragma unroll
  for (int mi = 0; mi < 2; ++mi) {
#pragma unroll
    for (int nt = 0; nt < 4; ++nt) {
#pragma unroll
      for (int r = 0; r < 8; ++r) {
        const int   sl = wm + mi * 16 + rb + r;
        const int   rg = r0 + wn + nt * 16 + cn;
        const float e  = __expf(acc[mi][nt][r] - 0.5f * ssq[sl]);
        ZP[((size_t)bh * Sq + s0 + sl) * Rf + rg] = (__bf16)e;
      }
    }
  }
}

// ---------------------------------------------------------------------------
// Kernel 3: ksum[bh][r] = sum_s k'[bh][s][r]  (split over S, f32 atomics)
// ---------------------------------------------------------------------------
__global__ __launch_bounds__(256) void ksum_kernel(
    const __bf16* __restrict__ KP, float* __restrict__ KS) {
  const int bh = blockIdx.x;
  const int s0 = blockIdx.y * (Sq / 8);
  const int r  = threadIdx.x;
  float acc = 0.0f;
  const __bf16* p = KP + ((size_t)bh * Sq + s0) * Rf + r;
  for (int s = 0; s < Sq / 8; ++s) acc += (float)p[(size_t)s * Rf];
  atomicAdd(&KS[bh * Rf + r], acc);
}

// ---------------------------------------------------------------------------
// Kernel 4: kvacc[bh][r][hd] += sum_s k'[s][r] * v[s][hd]  (WMMA, split-S)
// M=256 (8 waves x 32), N=64, K = 1024 per block (Sq/4 split).
// ---------------------------------------------------------------------------
#define KVP 40
__global__ __launch_bounds__(256) void kv_kernel(
    const __bf16* __restrict__ KP, const float* __restrict__ V,
    float* __restrict__ KVACC) {
  __shared__ __bf16 a_sh[256 * KVP];
  __shared__ __bf16 b_sh[64 * KVP];
  const int t   = threadIdx.x;
  const int wid = t >> 5;
  const int l   = t & 31;
  const int bh  = blockIdx.x;
  const int ks0 = blockIdx.y * (Sq / 4);
  const int wm  = wid * 32;

  v8f acc[2][4] = {};

  for (int s0 = ks0; s0 < ks0 + Sq / 4; s0 += 32) {
    {  // stage A = k'^T tile [256 r x 32 s] (transpose)
      const int sl = t & 31, rb0 = (t >> 5) * 32;
      const __bf16* p = KP + ((size_t)bh * Sq + s0 + sl) * Rf + rb0;
#pragma unroll
      for (int j = 0; j < 32; ++j) a_sh[(rb0 + j) * KVP + sl] = p[j];
    }
    {  // stage B = v^T tile [64 hd x 32 s] (f32 -> bf16, transpose)
      const int sl = t >> 3, hb = (t & 7) * 8;
      const float* p = V + ((size_t)bh * Sq + s0 + sl) * HDm + hb;
#pragma unroll
      for (int j = 0; j < 8; ++j) b_sh[(hb + j) * KVP + sl] = (__bf16)p[j];
    }
    __syncthreads();
    v16bf af[2];
#pragma unroll
    for (int mi = 0; mi < 2; ++mi) af[mi] = frag_a(a_sh + (wm + mi * 16) * KVP, KVP);
#pragma unroll
    for (int nt = 0; nt < 4; ++nt) {
      v16bf bf = frag_b(b_sh + nt * 16 * KVP, KVP);
#pragma unroll
      for (int mi = 0; mi < 2; ++mi) acc[mi][nt] = WMMA_BF16(af[mi], bf, acc[mi][nt]);
    }
    __syncthreads();
  }

  const int cn = l & 15;
  const int rb = (l < 16) ? 0 : 8;
#pragma unroll
  for (int mi = 0; mi < 2; ++mi)
#pragma unroll
    for (int nt = 0; nt < 4; ++nt)
#pragma unroll
      for (int r = 0; r < 8; ++r) {
        const int rg = wm + mi * 16 + rb + r;
        const int ng = nt * 16 + cn;
        atomicAdd(&KVACC[((size_t)bh * 256 + rg) * 64 + ng], acc[mi][nt][r]);
      }
}

// ---------------------------------------------------------------------------
// Kernel 5: finalize kv_ext[bh][r][0..79] bf16: cols 0..63 = kv', col 64 =
// ksum, cols 65..79 = 0 (norm rides along as an extra GEMM column).
// ---------------------------------------------------------------------------
__global__ __launch_bounds__(256) void kvfin_kernel(
    const float* __restrict__ KVACC, const float* __restrict__ KS,
    __bf16* __restrict__ KVE) {
  const int bh = blockIdx.x;
  const int r  = threadIdx.x;
  __bf16*      o = KVE + ((size_t)bh * 256 + r) * 80;
  const float* a = KVACC + ((size_t)bh * 256 + r) * 64;
#pragma unroll
  for (int c = 0; c < 64; ++c) o[c] = (__bf16)a[c];
  o[64] = (__bf16)KS[bh * Rf + r];
#pragma unroll
  for (int c = 65; c < 80; ++c) o[c] = (__bf16)0.0f;
}

// ---------------------------------------------------------------------------
// Kernel 6: attn = (q' @ kv_ext), normalize by column 64, scatter to [B,S,D].
// M=256 per block (8 waves x 32), N=80 (5 tiles), K=256.
// A tile is a pure bf16 copy -> async global->LDS when available.
// ---------------------------------------------------------------------------
#define AKP 40
__global__ __launch_bounds__(256) void attn_kernel(
    const __bf16* __restrict__ QP, const __bf16* __restrict__ KVE,
    float* __restrict__ O) {
  __shared__ __bf16 a_sh[256 * AKP];
  __shared__ __bf16 b_sh[80 * AKP];
  const int t   = threadIdx.x;
  const int wid = t >> 5;
  const int l   = t & 31;
  const int s0  = blockIdx.x * 256;
  const int bh  = blockIdx.y;
  const int b   = bh >> 4;
  const int h   = bh & 15;
  const int wm  = wid * 32;

  v8f acc[2][5] = {};

  for (int k0 = 0; k0 < Rf; k0 += 32) {
#if HAVE_ASYNC_LDS
    {  // stage A: 256 rows x 64B via GLOBAL_LOAD_ASYNC_TO_LDS_B128
       // each lane moves 16B to LDS addr row*AKP*2 + seg*2 (ASYNCcnt-tracked)
      const int rr  = t >> 2;          // 0..63
      const int seg = (t & 3) * 8;     // bf16 elems (16B)
#pragma unroll
      for (int i = 0; i < 4; ++i) {
        const int row = i * 64 + rr;
        const __bf16* g = QP + ((size_t)bh * Sq + s0 + row) * Rf + k0 + seg;
        __bf16*       s = a_sh + row * AKP + seg;
        __builtin_amdgcn_global_load_async_to_lds_b128((g_v4i_p)g, (l_v4i_p)s, 0, 0);
      }
    }
#else
    {  // stage A: row t, 32 contiguous k of q'
      const __bf16* p = QP + ((size_t)bh * Sq + s0 + t) * Rf + k0;
#pragma unroll
      for (int j = 0; j < 32; ++j) a_sh[t * AKP + j] = p[j];
    }
#endif
    {  // stage B: kv_ext^T tile [80 n x 32 k] (transpose)
      const int kk = t & 31, nb = (t >> 5) * 10;
      const __bf16* p = KVE + ((size_t)bh * 256 + k0 + kk) * 80 + nb;
#pragma unroll
      for (int j = 0; j < 10; ++j) b_sh[(nb + j) * AKP + kk] = p[j];
    }
#if HAVE_ASYNC_LDS
    __builtin_amdgcn_s_wait_asynccnt(0);
#endif
    __syncthreads();
    v16bf af[2];
#pragma unroll
    for (int mi = 0; mi < 2; ++mi) af[mi] = frag_a(a_sh + (wm + mi * 16) * AKP, AKP);
#pragma unroll
    for (int nt = 0; nt < 5; ++nt) {
      v16bf bf = frag_b(b_sh + nt * 16 * AKP, AKP);
#pragma unroll
      for (int mi = 0; mi < 2; ++mi) acc[mi][nt] = WMMA_BF16(af[mi], bf, acc[mi][nt]);
    }
    __syncthreads();
  }

  const int cn = l & 15;
  const int rb = (l < 16) ? 0 : 8;
#pragma unroll
  for (int mi = 0; mi < 2; ++mi) {
#pragma unroll
    for (int r = 0; r < 8; ++r) {
      // norm lives in n-tile 4 at local col 0 -> lane 0 (rows 0..7) / lane 16
      const float nv  = __shfl(acc[mi][4][r], l & 16, 32);
      const float inv = 1.0f / (nv + 1e-6f);
      const int   s   = s0 + wm + mi * 16 + rb + r;
#pragma unroll
      for (int nt = 0; nt < 4; ++nt)
        O[((size_t)b * Sq + s) * Dm + h * HDm + nt * 16 + cn] = acc[mi][nt][r] * inv;
    }
  }
}

// ---------------------------------------------------------------------------
// Host launcher
// ---------------------------------------------------------------------------
extern "C" void kernel_launch(void* const* d_in, const int* in_sizes, int n_in,
                              void* d_out, int out_size, void* d_ws, size_t ws_size,
                              hipStream_t stream) {
  (void)in_sizes; (void)n_in; (void)out_size; (void)ws_size;
  const float* x  = (const float*)d_in[0];
  const float* wq = (const float*)d_in[1];
  const float* bq = (const float*)d_in[2];
  const float* wk = (const float*)d_in[3];
  const float* bk = (const float*)d_in[4];
  const float* wv = (const float*)d_in[5];
  const float* bv = (const float*)d_in[6];
  const float* wo = (const float*)d_in[7];
  const float* bo = (const float*)d_in[8];
  const float* rf = (const float*)d_in[9];
  float* out = (float*)d_out;

  char*  ws  = (char*)d_ws;
  size_t off = 0;
  float*  q    = (float*)(ws + off); off += (size_t)BH * Sq * HDm * 4;  // 64 MB (reused for attn)
  float*  kbuf = (float*)(ws + off); off += (size_t)BH * Sq * HDm * 4;  // 64 MB
  float*  vbuf = (float*)(ws + off); off += (size_t)BH * Sq * HDm * 4;  // 64 MB
  __bf16* qp   = (__bf16*)(ws + off); off += (size_t)BH * Sq * Rf * 2;  // 128 MB
  __bf16* kp   = (__bf16*)(ws + off); off += (size_t)BH * Sq * Rf * 2;  // 128 MB
  float*  ksum = (float*)(ws + off); off += (size_t)BH * Rf * 4;        // 64 KB
  float*  kvac = (float*)(ws + off); off += (size_t)BH * 256 * 64 * 4;  // 4 MB
  __bf16* kve  = (__bf16*)(ws + off); off += (size_t)BH * 256 * 80 * 2; // 2.5 MB
  float*  attn = q;  // q is dead after feature_kernel; reuse as attn buffer

  // ksum and kvac are contiguous: zero both in one launch (atomics targets)
  const int nz = BH * Rf + BH * 256 * 64;
  zero_kernel<<<(nz + 255) / 256, 256, 0, stream>>>(ksum, nz);

  const dim3 gg(Bsz * Sq / 128, Dm / 128);
  gemm_bt_kernel<<<gg, 256, 0, stream>>>(x, wq, bq, q, 1);
  gemm_bt_kernel<<<gg, 256, 0, stream>>>(x, wk, bk, kbuf, 1);
  gemm_bt_kernel<<<gg, 256, 0, stream>>>(x, wv, bv, vbuf, 1);

  const dim3 gf(Sq / 128, Rf / 128, BH);
  feature_kernel<<<gf, 256, 0, stream>>>(q, rf, qp);
  feature_kernel<<<gf, 256, 0, stream>>>(kbuf, rf, kp);

  ksum_kernel<<<dim3(BH, 8), 256, 0, stream>>>(kp, ksum);
  kv_kernel<<<dim3(BH, 4), 256, 0, stream>>>(kp, vbuf, kvac);
  kvfin_kernel<<<BH, 256, 0, stream>>>(kvac, ksum, kve);

  attn_kernel<<<dim3(Sq / 256, BH), 256, 0, stream>>>(qp, kve, attn);

  gemm_bt_kernel<<<gg, 256, 0, stream>>>(attn, wo, bo, out, 0);
}
